// PolymerJEPAv2_39548058861723
// MI455X (gfx1250) — matrix operand order repaired and checked by reading, not compiled, not run on hardware
//
#include <hip/hip_runtime.h>
#include <hip/hip_bf16.h>

#define NNODES 200000
#define NEDGES 500000
#define NCTX   150000
#define ESUB   350000
#define NTGT   100000
#define NGRAPH 1024
#define FNODE  133
#define FEDGE  14
#define HDIM   128
#define TSTEPS 3

typedef __attribute__((ext_vector_type(16))) __bf16 v16bf;
typedef __attribute__((ext_vector_type(8)))  float  v8f;
typedef __attribute__((ext_vector_type(4)))  float  vf4;   // native clang vector (NT-load OK)

union BF16x16 {
  v16bf v;
  unsigned int u[8];
  uint4 q[2];
};

// epilogue / staging feature flags (compile-time)
enum : int {
  F_AG    = 1,   // gather A rows through an index map
  F_RES   = 2,   // add residual (h) before relu
  F_RELU  = 4,   // relu
  F_SCALE = 8,   // multiply rows by scale[]
  F_SG    = 16,  // scale[] indexed through sgather[]
  F_NT    = 32,  // non-temporal output stores (streaming operand)
};

__device__ __forceinline__ unsigned short f2bf(float f) {
  unsigned int u = __float_as_uint(f);
  u += 0x7FFFu + ((u >> 16) & 1u);   // round-to-nearest-even
  return (unsigned short)(u >> 16);
}

// Column-tile sweep + WMMA chain + epilogue. GUARD=false is the fast path
// taken by every full 128-row block (straight-line, no per-element EXEC ops).
template <int KPAD, int F, bool GUARD>
__device__ __forceinline__ void gemm_compute(
    const unsigned short* sW, const BF16x16* afr,
    long rowBase, int rLoc, int mloc, int hf,
    const float* __restrict__ bias,
    const float* __restrict__ res,
    const float* __restrict__ scale, const int* __restrict__ sgather,
    float* __restrict__ out, int n)
{
  constexpr int NCH = KPAD / 32;
  constexpr int STR = KPAD + 8;

  // Row scales depend only on the row -> hoist out of the column-tile loop.
  float rs[8];
  if constexpr (F & F_SCALE) {
#pragma unroll
    for (int v = 0; v < 8; ++v) {
      const long grow = rowBase + rLoc + hf * 8 + v;
      if (!GUARD || grow < n) {
        const long sr = (F & F_SG) ? (long)sgather[grow] : grow;
        rs[v] = scale[sr];
      } else {
        rs[v] = 0.0f;
      }
    }
  }

  for (int ct = 0; ct < 8; ++ct) {
    const int colg = ct * 16 + mloc;
    const float bia = bias[colg];
    // C/D 16x16 f32 layout: lanes 0-15 N=lane,M=v ; lanes 16-31 N=lane-16,M=v+8
    v8f c;
#pragma unroll
    for (int v = 0; v < 8; ++v) {
      const long grow = rowBase + rLoc + hf * 8 + v;
      float ci = 0.0f;
      if (!GUARD || grow < n) {
        ci = bia;
        if constexpr (F & F_RES) ci += res[grow * 128 + colg];
      }
      c[v] = ci;
    }
    const unsigned short* bp = sW + colg * STR + hf * 8;
#pragma unroll
    for (int ch = 0; ch < NCH; ++ch) {
      BF16x16 b;
      b.q[0] = *(const uint4*)(bp + ch * 32);       // k: kb..kb+7
      b.q[1] = *(const uint4*)(bp + ch * 32 + 16);  // k: 16+kb..+7
      c = __builtin_amdgcn_wmma_f32_16x16x32_bf16(false, afr[ch].v, false, b.v,
                                                  (short)0, c, false, false);
    }
#pragma unroll
    for (int v = 0; v < 8; ++v) {
      const long grow = rowBase + rLoc + hf * 8 + v;
      if (!GUARD || grow < n) {
        float o = c[v];
        if constexpr (F & F_RELU)  o = fmaxf(o, 0.0f);
        if constexpr (F & F_SCALE) o *= rs[v];
        float* p = out + grow * 128 + colg;
        if constexpr (F & F_NT) __builtin_nontemporal_store(o, p);
        else *p = o;
      }
    }
  }
}

// -------------------------------------------------------------------------
// WMMA GEMM: out[n x 128] = epi( A[gather][n x K] @ W[K x 128] )
// Wb: bf16, zero-padded to KPAD, transposed: Wb[c*KPAD + k].
// 256 threads = 8 wave32; wave owns a 16-row tile, iterates 8 col tiles.
// LDS rows padded +8 bf16 (4-bank skew) to kill bank conflicts.
// A fragments hoisted (loaded once, reused across all 8 col tiles).
// Fragments loaded as 16B uint4 -> ds_load_b128.
// -------------------------------------------------------------------------
template <int KPAD, int F>
__global__ void __launch_bounds__(256) gemm_bf16_wmma(
    const float* __restrict__ A, const int* __restrict__ gather, int K,
    const unsigned short* __restrict__ Wb,
    const float* __restrict__ bias,
    const float* __restrict__ res,
    const float* __restrict__ scale, const int* __restrict__ sgather,
    float* __restrict__ out, int n)
{
  constexpr int NCH = KPAD / 32;     // K chunks of 32
  constexpr int STR = KPAD + 8;      // padded LDS row stride (bf16 units)
  extern __shared__ unsigned short smem[];
  unsigned short* sW = smem;               // [128 cols][STR]
  unsigned short* sA = smem + 128 * STR;   // [128 rows][STR]
  const int tid = threadIdx.x;
  const long rowBase = (long)blockIdx.x * 128;

  // Stage weights: dense [c*KPAD+k] global -> padded LDS rows (u32 copies)
  {
    const unsigned int* s = (const unsigned int*)Wb;
    constexpr int U32_PER_ROW = KPAD / 2;
    for (int i = tid; i < 128 * U32_PER_ROW; i += 256) {
      const int c = i / U32_PER_ROW;
      const int r = i - c * U32_PER_ROW;
      ((unsigned int*)(sW + c * STR))[r] = s[i];
    }
  }
  // Stage activations: gather + f32->bf16 + zero-pad into padded rows
  for (int i = tid; i < 128 * KPAD; i += 256) {
    const int r = i / KPAD;
    const int k = i - r * KPAD;
    unsigned short v = 0;
    const long grow = rowBase + r;
    if (grow < n && k < K) {
      const long sr = (F & F_AG) ? (long)gather[grow] : grow;
      v = f2bf(A[sr * (long)K + k]);
    }
    sA[r * STR + k] = v;
  }
  __syncthreads();

  const int lane = tid & 31;
  const int wv   = tid >> 5;        // wave id -> row tile
  const int mloc = lane & 15;
  const int hf   = lane >> 4;       // lane half -> K sub-block
  const int rLoc = wv * 16;

  // Hoisted A fragments (ISA 7.12.2: 16-bit A 16x32 layout)
  const unsigned short* ap = sA + (rLoc + mloc) * STR;
  BF16x16 afr[NCH];
#pragma unroll
  for (int ch = 0; ch < NCH; ++ch) {
    afr[ch].q[0] = *(const uint4*)(ap + ch * 32 + hf * 8);       // k: kb..kb+7
    afr[ch].q[1] = *(const uint4*)(ap + ch * 32 + 16 + hf * 8);  // k: 16+kb..+7
  }

  if (rowBase + 128 <= n) {
    gemm_compute<KPAD, F, false>(sW, afr, rowBase, rLoc, mloc, hf,
                                 bias, res, scale, sgather, out, n);
  } else {
    gemm_compute<KPAD, F, true>(sW, afr, rowBase, rLoc, mloc, hf,
                                bias, res, scale, sgather, out, n);
  }
}

// Convert fp32 weight [K x 128] -> bf16 padded transposed [128 x Kpad]
__global__ void prep_w(const float* __restrict__ w, unsigned short* __restrict__ o,
                       int K, int Kpad) {
  const int i = blockIdx.x * 256 + threadIdx.x;
  if (i >= 128 * Kpad) return;
  const int c = i / Kpad;
  const int k = i - c * Kpad;
  o[i] = (k < K) ? f2bf(w[(size_t)k * HDIM + c]) : (unsigned short)0;
}

// cew[i] = edge_weight[sem[i]] * mask[i]
__global__ void cew_kernel(const float* __restrict__ ew, const int* __restrict__ sem,
                           const unsigned char* __restrict__ mask,
                           float* __restrict__ cew, int nES) {
  const int i = blockIdx.x * 256 + threadIdx.x;
  if (i < nES) cew[i] = ew[sem[i]] * (mask[i] ? 1.0f : 0.0f);
}

// One wave32 per edge: msg = ew*(h[src]+eh[e]); atomic scatter-add into agg[dst]
// eh is streamed (non-temporal); h/agg stay L2-resident.
__global__ void __launch_bounds__(256) msg_scatter(
    const float* __restrict__ h, const float* __restrict__ eh,
    const float* __restrict__ ew, const int* __restrict__ src,
    const int* __restrict__ dst, float* __restrict__ agg, int nE)
{
  const long t = (long)blockIdx.x * 256 + threadIdx.x;
  const long e = t >> 5;
  const int lane = (int)(t & 31);
  if (e >= nE) return;
  const int s = src[e];
  const int d = dst[e];
  const float w = ew[e];
  const vf4 hh = ((const vf4*)(h + (size_t)s * HDIM))[lane];
  const vf4 ee = __builtin_nontemporal_load(
      (const vf4*)(eh + (size_t)e * HDIM) + lane);
  float* ap = agg + (size_t)d * HDIM + lane * 4;
  atomicAdd(ap + 0, w * (hh.x + ee.x));
  atomicAdd(ap + 1, w * (hh.y + ee.y));
  atomicAdd(ap + 2, w * (hh.z + ee.z));
  atomicAdd(ap + 3, w * (hh.w + ee.w));
}

// segment sums: thread = (row, col). gatherVals=1 -> vals indexed by node.
__global__ void seg_scatter(const float* __restrict__ vals, const int* __restrict__ rowmap,
                            int gatherVals, const int* __restrict__ batch,
                            float* __restrict__ sum, float* __restrict__ cnt, int rows)
{
  const long t = (long)blockIdx.x * 256 + threadIdx.x;
  const long row = t >> 7;
  const int col = (int)(t & 127);
  if (row >= rows) return;
  const int node = rowmap[row];
  const int seg = batch[node];
  const float* vp = vals + (gatherVals ? (size_t)node : (size_t)row) * HDIM;
  atomicAdd(&sum[(size_t)seg * HDIM + col], vp[col]);
  if (col == 0) atomicAdd(&cnt[seg], 1.0f);
}

__global__ void finalize_kernel(const float* __restrict__ csum, const float* __restrict__ ccnt,
                                const float* __restrict__ tsum, const float* __restrict__ tcnt,
                                float* __restrict__ out) {
  const int i = blockIdx.x * 256 + threadIdx.x;
  if (i >= NGRAPH * HDIM) return;
  const int g = i >> 7;
  out[i] = csum[i] / fmaxf(ccnt[g], 1.0f);
  out[NGRAPH * HDIM + i] = tsum[i] / fmaxf(tcnt[g], 1.0f);
}

template <int KPAD, int F>
static void launch_gemm(const float* A, const int* g, int K, const unsigned short* W,
                        const float* b, const float* res,
                        const float* sc, const int* sg,
                        float* o, int n, hipStream_t stream) {
  dim3 grid((unsigned)((n + 127) / 128));
  size_t lds = (size_t)2 * 128 * (KPAD + 8) * sizeof(unsigned short);
  gemm_bf16_wmma<KPAD, F><<<grid, 256, lds, stream>>>(A, g, K, W, b, res, sc, sg, o, n);
}

extern "C" void kernel_launch(void* const* d_in, const int* in_sizes, int n_in,
                              void* d_out, int out_size, void* d_ws, size_t ws_size,
                              hipStream_t stream) {
  (void)in_sizes; (void)n_in; (void)out_size; (void)ws_size;
  const float* x           = (const float*)d_in[0];
  const float* node_weight = (const float*)d_in[1];
  const int*   edge_index  = (const int*)d_in[2];
  const float* edge_attr   = (const float*)d_in[3];
  const float* edge_weight = (const float*)d_in[4];
  const int*   batch       = (const int*)d_in[5];
  const int*   ctx_map     = (const int*)d_in[6];
  const int*   tgt_map     = (const int*)d_in[7];
  const int*   comb        = (const int*)d_in[8];
  const int*   sem         = (const int*)d_in[9];
  const unsigned char* cmask = (const unsigned char*)d_in[10];
  const float* P[16];
  for (int i = 0; i < 16; ++i) P[i] = (const float*)d_in[11 + i];
  // P[0..7]=ctx {w_node,b_node,w_edge,b_edge,w_mp,b_mp,w_out,b_out}, P[8..15]=tgt

  // ---- workspace carve ----
  char* ws = (char*)d_ws;
  size_t off = 0;
  auto carve = [&](size_t bytes) -> void* {
    void* p = ws + off;
    off += (bytes + 255) & ~(size_t)255;
    return p;
  };
  float* hbuf   = (float*)carve((size_t)NNODES * HDIM * 4);
  float* ehbuf  = (float*)carve((size_t)NEDGES * HDIM * 4);
  float* aggbuf = (float*)carve((size_t)NNODES * HDIM * 4);
  float* cew    = (float*)carve((size_t)ESUB * 4);
  float* csum   = (float*)carve((size_t)NGRAPH * HDIM * 4);
  float* tsum   = (float*)carve((size_t)NGRAPH * HDIM * 4);
  float* ccnt   = (float*)carve((size_t)NGRAPH * 4);
  float* tcnt   = (float*)carve((size_t)NGRAPH * 4);
  const int kpads[6] = {160, 32, 128, 128, 128, 128};  // node, edge, mp0-2, out
  unsigned short* wb[2][6];
  for (int p = 0; p < 2; ++p)
    for (int j = 0; j < 6; ++j)
      wb[p][j] = (unsigned short*)carve((size_t)128 * kpads[j] * 2);

  // ---- weight conversion (bf16, padded, transposed) ----
  for (int p = 0; p < 2; ++p) {
    prep_w<<<(128 * 160 + 255) / 256, 256, 0, stream>>>(P[p * 8 + 0], wb[p][0], FNODE, 160);
    prep_w<<<(128 * 32 + 255) / 256, 256, 0, stream>>>(P[p * 8 + 2], wb[p][1], FEDGE, 32);
    for (int t = 0; t < TSTEPS; ++t)
      prep_w<<<(128 * 128 + 255) / 256, 256, 0, stream>>>(
          P[p * 8 + 4] + (size_t)t * HDIM * HDIM, wb[p][2 + t], HDIM, 128);
    prep_w<<<(128 * 128 + 255) / 256, 256, 0, stream>>>(P[p * 8 + 6], wb[p][5], HDIM, 128);
  }

  (void)hipMemsetAsync(csum, 0, (size_t)NGRAPH * HDIM * 4, stream);
  (void)hipMemsetAsync(tsum, 0, (size_t)NGRAPH * HDIM * 4, stream);
  (void)hipMemsetAsync(ccnt, 0, (size_t)NGRAPH * 4, stream);
  (void)hipMemsetAsync(tcnt, 0, (size_t)NGRAPH * 4, stream);

  cew_kernel<<<(ESUB + 255) / 256, 256, 0, stream>>>(edge_weight, sem, cmask, cew, ESUB);

  // ================= context branch =================
  {
    const int n = NCTX, ne = ESUB;
    const int* esrc = comb;
    const int* edst = comb + ESUB;
    launch_gemm<160, F_AG | F_RELU>(x, ctx_map, FNODE, wb[0][0], P[1],
                                    nullptr, nullptr, nullptr, hbuf, n, stream);
    launch_gemm<32, F_AG | F_RELU | F_NT>(edge_attr, sem, FEDGE, wb[0][1], P[3],
                                          nullptr, nullptr, nullptr, ehbuf, ne, stream);
    for (int t = 0; t < TSTEPS; ++t) {
      (void)hipMemsetAsync(aggbuf, 0, (size_t)n * HDIM * 4, stream);
      msg_scatter<<<(ne + 7) / 8, 256, 0, stream>>>(hbuf, ehbuf, cew, esrc, edst, aggbuf, ne);
      launch_gemm<128, F_RES | F_RELU>(aggbuf, nullptr, HDIM, wb[0][2 + t],
                                       P[5] + (size_t)t * HDIM, hbuf,
                                       nullptr, nullptr, hbuf, n, stream);
    }
    launch_gemm<128, F_SCALE | F_SG>(hbuf, nullptr, HDIM, wb[0][5], P[7],
                                     nullptr, node_weight, ctx_map, aggbuf, n, stream);
    seg_scatter<<<(unsigned)(((size_t)n * 128 + 255) / 256), 256, 0, stream>>>(
        aggbuf, ctx_map, 0, batch, csum, ccnt, n);
  }
  // ================= target (full-graph) branch =================
  {
    const int n = NNODES, ne = NEDGES;
    const int* esrc = edge_index;
    const int* edst = edge_index + NEDGES;
    launch_gemm<160, F_RELU>(x, nullptr, FNODE, wb[1][0], P[9],
                             nullptr, nullptr, nullptr, hbuf, n, stream);
    launch_gemm<32, F_RELU | F_NT>(edge_attr, nullptr, FEDGE, wb[1][1], P[11],
                                   nullptr, nullptr, nullptr, ehbuf, ne, stream);
    for (int t = 0; t < TSTEPS; ++t) {
      (void)hipMemsetAsync(aggbuf, 0, (size_t)n * HDIM * 4, stream);
      msg_scatter<<<(ne + 7) / 8, 256, 0, stream>>>(hbuf, ehbuf, edge_weight, esrc, edst,
                                                    aggbuf, ne);
      launch_gemm<128, F_RES | F_RELU>(aggbuf, nullptr, HDIM, wb[1][2 + t],
                                       P[13] + (size_t)t * HDIM, hbuf,
                                       nullptr, nullptr, hbuf, n, stream);
    }
    launch_gemm<128, F_SCALE>(hbuf, nullptr, HDIM, wb[1][5], P[15],
                              nullptr, node_weight, nullptr, aggbuf, n, stream);
    seg_scatter<<<(unsigned)(((size_t)NTGT * 128 + 255) / 256), 256, 0, stream>>>(
        aggbuf, tgt_map, 1, batch, tsum, tcnt, NTGT);
  }

  finalize_kernel<<<(NGRAPH * HDIM + 255) / 256, 256, 0, stream>>>(
      csum, ccnt, tsum, tcnt, (float*)d_out);
}
